// CIFAR10_Monarch_MLP_30709016166960
// MI455X (gfx1250) — compile-verified, matrix operand order checked by reading, not run on hardware
//
#include <hip/hip_runtime.h>
#include <hip/hip_bf16.h>

typedef __attribute__((ext_vector_type(16))) __bf16 v16bf;
typedef __attribute__((ext_vector_type(8)))  __bf16 v8bf;
typedef __attribute__((ext_vector_type(8)))  float  v8f;

#define BM 32            // batch rows per workgroup (two 16-row M-tiles)
#define NTHREADS 512     // 16 waves

__device__ __forceinline__ v16bf make_frag(const __bf16* lo, const __bf16* hi) {
    union { v16bf v; v8bf h[2]; } u;
    u.h[0] = *(const v8bf*)lo;   // VGPR0-3: K = base .. base+7
    u.h[1] = *(const v8bf*)hi;   // VGPR4-7: K = base+16 .. base+23
    return u.v;
}

// ---------------------------------------------------------------------------
// Fused Monarch layer: GEMM1 (block-diag) -> butterfly permute (in LDS) ->
// GEMM2 (block-diag) -> optional ReLU. bf16 in / fp32 accumulate.
// Weights pre-padded: w1p (k, QPAD, PPAD), w2p (l, SPAD, RPAD), zero-filled.
// Each wave computes TWO 16x16 M-tiles per column tile: one B fragment feeds
// two v_wmma, halving weight (L2) traffic versus BM=16.
// ---------------------------------------------------------------------------
template<bool RELU, bool OUT_F32>
__global__ void __launch_bounds__(NTHREADS)
monarch_layer(const __bf16* __restrict__ hin,
              const __bf16* __restrict__ w1p,
              const __bf16* __restrict__ w2p,
              __bf16* __restrict__ hout_bf,
              float*  __restrict__ hout_f,
              int k, int q, int p, int l, int s, int r,
              int in_f, int out_f,
              int QPAD, int PPAD, int SPAD, int RPAD)
{
    extern __shared__ __align__(16) char smem_raw[];
    const int xsPitch  = PPAD + 8;          // bf16 elements, 16B-aligned rows
    const int midPitch = l * RPAD + 8;
    __bf16* xs   = (__bf16*)smem_raw;                 // BM x xsPitch
    __bf16* mids = xs + BM * xsPitch;                 // BM x midPitch (permuted o1)

    const int tid   = threadIdx.x;
    const int lane  = tid & 31;
    const int wave  = tid >> 5;
    const int nwave = blockDim.x >> 5;
    const int b0    = blockIdx.x * BM;

    const int mrow  = lane & 15;            // A row / B column within 16x16 tile
    const int koff  = (lane >> 4) << 3;     // 0 for lanes 0-15, 8 for 16-31
    const int rbase = (lane >> 4) << 3;     // C/D: rows j (+8 for upper half-wave)

    // Zero all of LDS (covers K-pad and R-pad regions read by WMMA).
    {
        unsigned int* z = (unsigned int*)smem_raw;
        const int n32 = (BM * (xsPitch + midPitch)) >> 1;
        for (int i = tid; i < n32; i += blockDim.x) z[i] = 0u;
    }

    // ---------------- GEMM1 over the k diagonal blocks ----------------
    const int nt1 = QPAD >> 4;
    for (int kb = 0; kb < k; ++kb) {
        __syncthreads();                    // xs free (also fences the zeroing)
        if ((p & 1) == 0) {                 // vectorized u32 staging path
            const int pw = p >> 1;
            for (int i = tid; i < BM * pw; i += blockDim.x) {
                int row = i / pw, cw = i - row * pw;
                ((unsigned int*)(xs + row * xsPitch))[cw] =
                    ((const unsigned int*)(hin + (size_t)(b0 + row) * in_f + (size_t)kb * p))[cw];
            }
        } else {
            for (int i = tid; i < BM * p; i += blockDim.x) {
                int row = i / p, col = i - row * p;
                xs[row * xsPitch + col] = hin[(size_t)(b0 + row) * in_f + (size_t)kb * p + col];
            }
        }
        __syncthreads();                    // xs ready

        const __bf16* w1b = w1p + (size_t)kb * QPAD * PPAD;
        for (int qt = wave; qt < nt1; qt += nwave) {
            v8f acc0 = {}, acc1 = {};
            const __bf16* arow0 = xs  + mrow * xsPitch;
            const __bf16* arow1 = xs  + (mrow + 16) * xsPitch;
            const __bf16* brow  = w1b + (size_t)(qt * 16 + mrow) * PPAD;
            for (int kk = 0; kk < PPAD; kk += 32) {
                v16bf b  = make_frag(brow  + kk + koff, brow  + kk + koff + 16);
                v16bf a0 = make_frag(arow0 + kk + koff, arow0 + kk + koff + 16);
                v16bf a1 = make_frag(arow1 + kk + koff, arow1 + kk + koff + 16);
                acc0 = __builtin_amdgcn_wmma_f32_16x16x32_bf16(
                           false, a0, false, b, (short)0, acc0, false, false);
                acc1 = __builtin_amdgcn_wmma_f32_16x16x32_bf16(
                           false, a1, false, b, (short)0, acc1, false, false);
            }
            // Butterfly permutation: flat = kb*q + qi ; (l,r) = (flat%l, flat/l)
            const int qi = qt * 16 + mrow;
            if (qi < q) {
                const int flat = kb * q + qi;
                const int li = flat % l;
                const int ri = flat / l;
                const int colAddr = li * RPAD + ri;
#pragma unroll
                for (int j = 0; j < 8; ++j) {
                    mids[(j + rbase) * midPitch + colAddr]        = (__bf16)acc0[j];
                    mids[(j + rbase + 16) * midPitch + colAddr]   = (__bf16)acc1[j];
                }
            }
        }
    }
    __syncthreads();                        // mids fully written (permuted o1)

    // ---------------- GEMM2 over the l diagonal blocks ----------------
    const int nt2 = SPAD >> 4;
    const int T = l * nt2;
    for (int t = wave; t < T; t += nwave) {
        const int lb = t / nt2;
        const int st = t - lb * nt2;
        v8f acc0 = {}, acc1 = {};
        const __bf16* arow0 = mids + mrow * midPitch + lb * RPAD;
        const __bf16* arow1 = mids + (mrow + 16) * midPitch + lb * RPAD;
        const __bf16* brow  = w2p + (size_t)(lb * SPAD + st * 16 + mrow) * RPAD;
        for (int kk = 0; kk < RPAD; kk += 32) {
            v16bf b  = make_frag(brow  + kk + koff, brow  + kk + koff + 16);
            v16bf a0 = make_frag(arow0 + kk + koff, arow0 + kk + koff + 16);
            v16bf a1 = make_frag(arow1 + kk + koff, arow1 + kk + koff + 16);
            acc0 = __builtin_amdgcn_wmma_f32_16x16x32_bf16(
                       false, a0, false, b, (short)0, acc0, false, false);
            acc1 = __builtin_amdgcn_wmma_f32_16x16x32_bf16(
                       false, a1, false, b, (short)0, acc1, false, false);
        }
        const int si = st * 16 + mrow;
        if (si < s) {
            const int col = lb * s + si;
            if (col < out_f) {
#pragma unroll
                for (int j = 0; j < 8; ++j) {
                    float v0 = acc0[j], v1 = acc1[j];
                    if (RELU) { v0 = v0 > 0.f ? v0 : 0.f; v1 = v1 > 0.f ? v1 : 0.f; }
                    if (OUT_F32) {
                        hout_f[(size_t)(b0 + j + rbase) * out_f + col]      = v0;
                        hout_f[(size_t)(b0 + j + rbase + 16) * out_f + col] = v1;
                    } else {
                        hout_bf[(size_t)(b0 + j + rbase) * out_f + col]      = (__bf16)v0;
                        hout_bf[(size_t)(b0 + j + rbase + 16) * out_f + col] = (__bf16)v1;
                    }
                }
            }
        }
    }
}

// -------------------------- prep / epilogue kernels --------------------------
__global__ void cvt_f32_bf16(const float* __restrict__ in, __bf16* __restrict__ out, size_t n) {
    size_t i = (size_t)blockIdx.x * blockDim.x + threadIdx.x;
    if (i < n) out[i] = (__bf16)in[i];
}

__global__ void pad_weight(const float* __restrict__ in, __bf16* __restrict__ out,
                           int nb, int rows, int cols, int rpad, int cpad) {
    size_t i = (size_t)blockIdx.x * blockDim.x + threadIdx.x;
    size_t total = (size_t)nb * rpad * cpad;
    if (i >= total) return;
    int cc = (int)(i % cpad);
    size_t t = i / cpad;
    int rr = (int)(t % rpad);
    int bb = (int)(t / rpad);
    float v = 0.f;
    if (rr < rows && cc < cols) v = in[((size_t)bb * rows + rr) * cols + cc];
    out[i] = (__bf16)v;
}

__global__ void log_softmax10(const float* __restrict__ logits, float* __restrict__ out, int nrows) {
    int row = blockIdx.x * blockDim.x + threadIdx.x;
    if (row >= nrows) return;
    const float* x = logits + (size_t)row * 10;
    float m = x[0];
#pragma unroll
    for (int j = 1; j < 10; ++j) m = fmaxf(m, x[j]);
    float ssum = 0.f;
#pragma unroll
    for (int j = 0; j < 10; ++j) ssum += expf(x[j] - m);
    float lg = logf(ssum);
    float* o = out + (size_t)row * 10;
#pragma unroll
    for (int j = 0; j < 10; ++j) o[j] = x[j] - m - lg;
}

// ------------------------------- host driver --------------------------------
static inline int al16(int v) { return (v + 15) & ~15; }
static inline int al32(int v) { return (v + 31) & ~31; }

extern "C" void kernel_launch(void* const* d_in, const int* in_sizes, int n_in,
                              void* d_out, int out_size, void* d_ws, size_t ws_size,
                              hipStream_t stream) {
    (void)in_sizes; (void)n_in; (void)out_size; (void)ws_size;
    const int BATCH = 8192, IN_F = 3072;
    struct LD { int k, q, p, l, s, r, out; };
    static const LD L[7] = {
        {4, 750, 768, 4, 750, 750, 3000},
        {4, 500, 750, 4, 500, 500, 2000},
        {4, 250, 500, 4, 250, 250, 1000},
        {4, 125, 250, 4, 125, 125,  500},
        {4,  50, 125, 4,  50,  50,  200},
        {4,  25,  50, 4,  25,  25,  100},
        {4,   3,  25, 4,   3,   3,   10}};

    char* ws = (char*)d_ws;
    size_t off = 0;
    auto carve = [&](size_t bytes) -> char* {
        char* ptr = ws + off;
        off = (off + bytes + 255) & ~(size_t)255;
        return ptr;
    };

    __bf16* xbf = (__bf16*)carve((size_t)BATCH * IN_F * 2);
    __bf16* w1p[7]; __bf16* w2p[7];
    int QP[7], PP[7], SP[7], RP[7];
    for (int i = 0; i < 7; ++i) {
        QP[i] = al16(L[i].q); PP[i] = al32(L[i].p);
        SP[i] = al16(L[i].s); RP[i] = al32(L[i].r);
        w1p[i] = (__bf16*)carve((size_t)L[i].k * QP[i] * PP[i] * 2);
        w2p[i] = (__bf16*)carve((size_t)L[i].l * SP[i] * RP[i] * 2);
    }
    __bf16* hA = (__bf16*)carve((size_t)BATCH * 3000 * 2);
    __bf16* hB = (__bf16*)carve((size_t)BATCH * 3000 * 2);
    float* logits = (float*)carve((size_t)BATCH * 10 * 4);

    // One-time conversions (run every call; deterministic).
    {
        size_t nx = (size_t)BATCH * IN_F;
        cvt_f32_bf16<<<dim3((unsigned)((nx + 255) / 256)), dim3(256), 0, stream>>>(
            (const float*)d_in[0], xbf, nx);
        for (int i = 0; i < 7; ++i) {
            size_t t1 = (size_t)L[i].k * QP[i] * PP[i];
            pad_weight<<<dim3((unsigned)((t1 + 255) / 256)), dim3(256), 0, stream>>>(
                (const float*)d_in[1 + 2 * i], w1p[i], L[i].k, L[i].q, L[i].p, QP[i], PP[i]);
            size_t t2 = (size_t)L[i].l * SP[i] * RP[i];
            pad_weight<<<dim3((unsigned)((t2 + 255) / 256)), dim3(256), 0, stream>>>(
                (const float*)d_in[2 + 2 * i], w2p[i], L[i].l, L[i].s, L[i].r, SP[i], RP[i]);
        }
    }

    // Layers 1-2 need large dynamic LDS (full-width permuted intermediate):
    // layer 1 uses 32*(776 + 3080)*2 = 246,784 B <= 320 KB/WGP workgroup limit.
    (void)hipFuncSetAttribute((const void*)monarch_layer<true, false>,
                              hipFuncAttributeMaxDynamicSharedMemorySize, 253952);

    const __bf16* hin = xbf;
    int in_f = IN_F;
    const dim3 grid(BATCH / BM), block(NTHREADS);
    for (int i = 0; i < 7; ++i) {
        size_t lds = (size_t)BM * ((PP[i] + 8) + (L[i].l * RP[i] + 8)) * 2;
        if (i < 6) {
            __bf16* hout = (i & 1) ? hB : hA;
            monarch_layer<true, false><<<grid, block, lds, stream>>>(
                hin, w1p[i], w2p[i], hout, nullptr,
                L[i].k, L[i].q, L[i].p, L[i].l, L[i].s, L[i].r,
                in_f, L[i].out, QP[i], PP[i], SP[i], RP[i]);
            hin = hout;
            in_f = L[i].out;
        } else {
            monarch_layer<false, true><<<grid, block, lds, stream>>>(
                hin, w1p[i], w2p[i], nullptr, logits,
                L[i].k, L[i].q, L[i].p, L[i].l, L[i].s, L[i].r,
                in_f, L[i].out, QP[i], PP[i], SP[i], RP[i]);
        }
    }

    log_softmax10<<<dim3(BATCH / 256), dim3(256), 0, stream>>>(logits, (float*)d_out, BATCH);
}
